// SpectralConv1d_51376398795175
// MI455X (gfx1250) — matrix operand order repaired and checked by reading, not compiled, not run on hardware
//
#include <hip/hip_runtime.h>

// SpectralConv1d (FNO) for MI455X / gfx1250, wave32 + WMMA f16 + async-to-LDS.
// Only 64 modes are live -> 3 small GEMMs:
//   S1: X[2048,128]   = x[2048,4096] @ cftB[4096,128]      (re/im interleaved cols)
//   S2: low[2048,128] = per-f complex mix, [32x128]@[128x128] per frequency
//   S3: out[2048,4096]= low[2048,128] @ icftB[128,4096]
// Bases are f16 generated on-device in FRAGMENT-READY order:
//   dword[(kstep16)*NC*16 + col*16 + q] = K-pair (2*(kstep16*16+q), +1) at `col`
// so a wave's B fragment is two b128 loads per lane.

typedef __attribute__((ext_vector_type(16))) _Float16 v16h;
typedef __attribute__((ext_vector_type(8)))  float    v8f;
typedef __attribute__((ext_vector_type(4)))  unsigned v4u;
typedef __attribute__((ext_vector_type(4)))  int      v4i;

#define N_TIME 4096
#define KC     128          // packed re/im column count (2*modes)
#define TWO_PI_OVER_N 1.5339807878856412e-03f   // 2*pi/4096

// ---- CDNA5 async global->LDS copy (guarded; falls back to manual copy) ----
#if defined(__has_builtin)
# if __has_builtin(__builtin_amdgcn_global_load_async_to_lds_b128)
#  define HAVE_ASYNC_TO_LDS 1
# endif
#endif
#ifndef HAVE_ASYNC_TO_LDS
# define HAVE_ASYNC_TO_LDS 0
#endif

#if HAVE_ASYNC_TO_LDS
typedef __attribute__((address_space(1))) v4i* gv4i_p;   // global (AS1)
typedef __attribute__((address_space(3))) v4i* lv4i_p;   // LDS (AS3)
#endif

__device__ __forceinline__ void copy16B_to_lds(const float* gsrc, float* ldst) {
#if HAVE_ASYNC_TO_LDS
    gv4i_p g = (gv4i_p)(unsigned long long)(uintptr_t)gsrc;  // flat VA == global VA
    lv4i_p l = (lv4i_p)ldst;                                 // addrspacecast -> LDS
    __builtin_amdgcn_global_load_async_to_lds_b128(g, l, 0, 0);
#else
    *(float4*)ldst = *(const float4*)gsrc;
#endif
}

__device__ __forceinline__ void wait_stage() {
#if HAVE_ASYNC_TO_LDS
# if defined(__has_builtin) && __has_builtin(__builtin_amdgcn_s_wait_asynccnt)
    __builtin_amdgcn_s_wait_asynccnt(0);
# else
    asm volatile("s_wait_asynccnt 0x0" ::: "memory");
# endif
#endif
}

__device__ __forceinline__ unsigned pack2h(float a, float b) {
    union { _Float16 h[2]; unsigned u; } p;
    p.h[0] = (_Float16)a; p.h[1] = (_Float16)b; return p.u;
}

__device__ __forceinline__ v16h frag_from2(v4u lo, v4u hi) {
    union { v4u u[2]; v16h h; } x; x.u[0] = lo; x.u[1] = hi; return x.h;
}

// A fragment built from a RAW f32 LDS tile (convert while building).
// VGPR v holds K pair k0 = 32*s + (lane/16)*16 + v*2.
__device__ __forceinline__ v16h fragA_f32(const float* lds, int rowStride,
                                          int m, int s, int lh) {
    union { unsigned u[8]; v16h h; } x;
#pragma unroll
    for (int v = 0; v < 8; ++v) {
        int k0 = s * 32 + lh * 16 + v * 2;
        float2 ab = *(const float2*)(lds + m * rowStride + k0);
        x.u[v] = pack2h(ab.x, ab.y);
    }
    return x.h;
}

// B fragment from a fragment-ready global block: 2 x b128 per lane.
__device__ __forceinline__ v16h fragB_fr(const unsigned* __restrict__ blk,
                                         int col, int lh) {
    const v4u* p = (const v4u*)(blk + col * 16 + lh * 8);
    return frag_from2(p[0], p[1]);
}

// Packed-f16 LDS fragment helpers (stage 2 only).
__device__ __forceinline__ v16h lds_fragA_pk(const unsigned* lds, int rowStride,
                                             int m, int s, int lh) {
    int base = m * rowStride + s * 16 + lh * 4;
    return frag_from2(*(const v4u*)(lds + base), *(const v4u*)(lds + base + 8));
}
__device__ __forceinline__ v16h lds_fragB_pk(const unsigned* lds, int kPairs,
                                             int col, int s, int lh) {
    int base = col * kPairs + s * 16 + lh * 4;
    return frag_from2(*(const v4u*)(lds + base), *(const v4u*)(lds + base + 8));
}

// ---------------- basis generation (fragment-ready layouts) ----------------
// cftB: blocks of 16 K-pairs. dword[blk*2048 + col*16 + q] = (e(t0,col), e(t0+1,col)),
// t0 = 2*(blk*16+q); col 2f -> cos(2*pi*f*t/n), col 2f+1 -> -sin.
__global__ void gen_cft_basis(unsigned* __restrict__ cftB) {
    int idx = blockIdx.x * blockDim.x + threadIdx.x;   // 0..262143
    int q   = idx & 15;
    int col = (idx >> 4) & 127;
    int blk = idx >> 11;
    int p   = blk * 16 + q;                            // global K pair -> t pair
    int f   = col >> 1;
    float vals[2];
#pragma unroll
    for (int e = 0; e < 2; ++e) {
        int t  = 2 * p + e;
        int ph = (f * t) & (N_TIME - 1);               // exact phase mod n
        float s, c;
        __sincosf((float)ph * TWO_PI_OVER_N, &s, &c);
        vals[e] = (col & 1) ? -s : c;
    }
    cftB[idx] = pack2h(vals[0], vals[1]);
}

// icftB: dword[s*65536 + t*16 + q] = (cos phi, -sin phi), f = s*16+q,
// phi = 2*pi*t*(f-2048)/n  (fftshift pairing from the reference).
__global__ void gen_icft_basis(unsigned* __restrict__ icftB) {
    int idx = blockIdx.x * blockDim.x + threadIdx.x;   // 0..262143
    int q = idx & 15;
    int t = (idx >> 4) & (N_TIME - 1);
    int s = idx >> 16;
    int f = s * 16 + q;
    int ph = (t * (f - 2048)) & (N_TIME - 1);          // low bits == mod 4096
    float sn, cs;
    __sincosf((float)ph * TWO_PI_OVER_N, &sn, &cs);
    icftB[idx] = pack2h(cs, -sn);
}

// ---------------- stage 1: CFT GEMM [2048,4096]x[4096,128] ----------------
// K-tile 64 (2 WMMA / barrier), double-buffered async A staging.
__global__ __launch_bounds__(128) void cft_gemm(const float* __restrict__ x,
                                                const unsigned* __restrict__ cftB,
                                                float* __restrict__ X) {
    __shared__ __attribute__((aligned(16))) float Abuf[2][16 * 64];  // 2 x 4 KB raw f32
    const int tid  = threadIdx.x;
    const int wave = tid >> 5, lane = tid & 31;
    const int lh = lane >> 4, ln = lane & 15;
    const int mt = blockIdx.x;                         // 0..127 (M tile)
    const int c0 = blockIdx.y * 64;                    // col half
    const int col = c0 + wave * 16 + ln;

    auto stageA = [&](int buf, int ks) {
#pragma unroll
        for (int e = 0; e < 2; ++e) {
            int fi = tid + e * 128;                    // float4 index, 256 total
            int r  = fi >> 4;                          // 16 float4 per 64-float row
            int kq = (fi & 15) * 4;
            copy16B_to_lds(x + (mt * 16 + r) * N_TIME + ks * 64 + kq,
                           &Abuf[buf][r * 64 + kq]);
        }
    };

    stageA(0, 0);
    wait_stage();
    __syncthreads();

    v8f acc = {};
    for (int ks = 0; ks < 64; ++ks) {
        int cur = ks & 1;
        if (ks < 63) stageA(cur ^ 1, ks + 1);
        __builtin_prefetch(cftB + (2 * ks + 2) * 2048 + col * 16, 0, 1);
#pragma unroll
        for (int ss = 0; ss < 2; ++ss) {
            v16h af = fragA_f32(&Abuf[cur][0], 64, ln, ss, lh);
            v16h bf = fragB_fr(cftB + (2 * ks + ss) * 2048, col, lh);
            acc = __builtin_amdgcn_wmma_f32_16x16x32_f16(false, af, false, bf,
                                                         (short)0, acc, false, false);
        }
        wait_stage();
        __syncthreads();
    }
#pragma unroll
    for (int v = 0; v < 8; ++v)
        X[(mt * 16 + v + 8 * lh) * KC + col] = acc[v];
}

// ---------------- stage 2: per-frequency complex mix ----------------
// One block per f. low_re = Xre*Wre - Xim*Wim ; low_im = Xre*Wim + Xim*Wre (dt folded).
__global__ __launch_bounds__(128) void mode_mix(const float* __restrict__ X,
                                                const float* __restrict__ wre,
                                                const float* __restrict__ wim,
                                                float* __restrict__ low) {
    __shared__ unsigned A2[32 * 64];                   // [b][kpair] 8 KB
    __shared__ unsigned B2[128 * 64];                  // [o'][kpair] 32 KB
    const int f = blockIdx.x;
    const int tid = threadIdx.x;
    const float dt = 1.0f / (float)N_TIME;

#pragma unroll 4
    for (int i = 0; i < 16; ++i) {                     // A2: stack Xre(k<64) / Xim
        int lin = tid + i * 128;
        int b = lin >> 6, kp = lin & 63, k = kp * 2;
        float v0 = (k < 64)  ? X[(b * 64 + k) * KC + 2 * f]
                             : X[(b * 64 + (k - 64)) * KC + 2 * f + 1];
        int k1 = k + 1;
        float v1 = (k1 < 64) ? X[(b * 64 + k1) * KC + 2 * f]
                             : X[(b * 64 + (k1 - 64)) * KC + 2 * f + 1];
        A2[lin] = pack2h(v0, v1);
    }
#pragma unroll 4
    for (int i = 0; i < 64; ++i) {                     // B2 col-major: [Wre;-Wim]|[Wim;Wre]
        int lin = tid + i * 128;
        int op = lin >> 6, kp = lin & 63;
        int o = op >> 1, rr = op & 1;
        float v[2];
#pragma unroll
        for (int e = 0; e < 2; ++e) {
            int kk = kp * 2 + e;
            int ii = (kk < 64) ? kk : kk - 64;
            float a  = wre[(ii * 64 + o) * 64 + f];
            float bb = wim[(ii * 64 + o) * 64 + f];
            v[e] = rr == 0 ? ((kk < 64) ? a : -bb) : ((kk < 64) ? bb : a);
        }
        B2[op * 64 + kp] = pack2h(v[0] * dt, v[1] * dt);
    }
    __syncthreads();

    const int wave = tid >> 5, lane = tid & 31;
    const int lh = lane >> 4, ln = lane & 15;
    for (int tI = 0; tI < 4; ++tI) {                   // 16 tiles / 4 waves
        int tile = wave * 4 + tI;
        int mT = tile >> 3, nT = tile & 7;
        v8f acc = {};
#pragma unroll
        for (int s = 0; s < 4; ++s) {
            v16h af = lds_fragA_pk(A2, 64, mT * 16 + ln, s, lh);
            v16h bf = lds_fragB_pk(B2, 64, nT * 16 + ln, s, lh);
            acc = __builtin_amdgcn_wmma_f32_16x16x32_f16(false, af, false, bf,
                                                         (short)0, acc, false, false);
        }
#pragma unroll
        for (int v = 0; v < 8; ++v) {
            int b  = mT * 16 + v + 8 * lh;
            int op = nT * 16 + ln;
            low[(b * 64 + (op >> 1)) * KC + 2 * f + (op & 1)] = acc[v];
        }
    }
}

// ---------------- stage 3: ICFT GEMM [2048,128]x[128,4096] ----------------
__global__ __launch_bounds__(128) void icft_gemm(const float* __restrict__ low,
                                                 const unsigned* __restrict__ icftB,
                                                 float* __restrict__ out) {
    __shared__ __attribute__((aligned(16))) float A3[16 * 128];   // 8 KB raw f32
    const int tid = threadIdx.x, wave = tid >> 5, lane = tid & 31;
    const int lh = lane >> 4, ln = lane & 15;
    const int mt  = blockIdx.x;                        // 0..127
    const int t0b = blockIdx.y * 256;                  // 256 t-cols per block

#pragma unroll
    for (int q = 0; q < 4; ++q) {                      // async-stage full 16x128 A tile
        int fi = tid + q * 128;                        // float4 index, 512 total
        int r  = fi >> 5;                              // 32 float4 per 128-float row
        int kq = (fi & 31) * 4;
        copy16B_to_lds(low + (mt * 16 + r) * KC + kq, &A3[r * 128 + kq]);
    }
    wait_stage();
    __syncthreads();

    for (int g = 0; g < 4; ++g) {
        int t0  = t0b + g * 64;
        int col = t0 + wave * 16 + ln;
        __builtin_prefetch(icftB + col * 16 + 1024, 0, 1);
        v8f acc = {};
#pragma unroll
        for (int s = 0; s < 4; ++s) {
            v16h af = fragA_f32(A3, 128, ln, s, lh);
            v16h bf = fragB_fr(icftB + s * 65536, col, lh);
            acc = __builtin_amdgcn_wmma_f32_16x16x32_f16(false, af, false, bf,
                                                         (short)0, acc, false, false);
        }
#pragma unroll
        for (int v = 0; v < 8; ++v)
            out[(mt * 16 + v + 8 * lh) * N_TIME + col] = acc[v];
    }
}

extern "C" void kernel_launch(void* const* d_in, const int* in_sizes, int n_in,
                              void* d_out, int out_size, void* d_ws, size_t ws_size,
                              hipStream_t stream) {
    const float* x   = (const float*)d_in[0];   // [32,64,4096]
    const float* wre = (const float*)d_in[1];   // [64,64,64]
    const float* wim = (const float*)d_in[2];   // [64,64,64]
    float* out = (float*)d_out;                 // [32,64,4096]

    // workspace layout (4 MB total): cftB | icftB | X | low
    unsigned* cftB  = (unsigned*)d_ws;                // 2048*128 dwords (1 MB)
    unsigned* icftB = cftB + 2048 * KC;               // 64*4096 dwords (1 MB)
    float*    X     = (float*)(icftB + 64 * N_TIME);  // [2048][128] f32 (1 MB)
    float*    low   = X + 2048 * KC;                  // [2048][128] f32 (1 MB)

    gen_cft_basis <<<1024, 256, 0, stream>>>(cftB);
    gen_icft_basis<<<1024, 256, 0, stream>>>(icftB);
    cft_gemm <<<dim3(128, 2),  128, 0, stream>>>(x, cftB, X);
    mode_mix <<<64,            128, 0, stream>>>(X, wre, wim, low);
    icft_gemm<<<dim3(128, 16), 128, 0, stream>>>(low, icftB, out);
}
